// MinkUNetBase_38955353374932
// MI455X (gfx1250) — compile-verified, matrix operand order checked by reading, not run on hardware
//
#include <hip/hip_runtime.h>

// ---------------------------------------------------------------------------
// CDNA5 (gfx1250) MinkUNet forward. All GEMMs via V_WMMA_F32_16X16X32_BF16,
// prepacked bf16 weight fragments, b128 gathered activations with software-
// pipelined global_prefetch of the next tap's gather, multi-tile accumulators
// per wave, and BN statistics fused into the conv epilogue.
// ---------------------------------------------------------------------------

typedef __attribute__((ext_vector_type(16))) __bf16 v16bf;
typedef __attribute__((ext_vector_type(8)))  float  v8f;

union BF16Frag { v16bf v; unsigned short h[16]; unsigned int u[8]; };

__device__ __forceinline__ unsigned short f2bf(float f) {
    unsigned int u = __float_as_uint(f);
    u += 0x7FFFu + ((u >> 16) & 1u);      // round-to-nearest-even
    return (unsigned short)(u >> 16);
}

// ------------------------------ utility kernels ----------------------------

__global__ void k_zero(float* __restrict__ p, long long n) {
    long long i = (long long)blockIdx.x * blockDim.x + threadIdx.x;
    if (i < n) p[i] = 0.f;
}

// feats (N x Csrc fp32) -> padded bf16 (N x 32), channels >= Csrc zeroed
__global__ void k_cvt_pad(const float* __restrict__ s, unsigned short* __restrict__ d,
                          long long n /* = N*32 */, int Csrc) {
    long long i = (long long)blockIdx.x * blockDim.x + threadIdx.x;
    if (i >= n) return;
    int c = (int)(i & 31);
    long long r = i >> 5;
    d[i] = (c < Csrc) ? f2bf(s[r * Csrc + c]) : (unsigned short)0;
}

// Repack fp32 weights [T][CiSrc][Co] into bf16 WMMA B fragments:
//   Wf[((t*KC + kc)*NT + nt)*32 + lane][16], KC = CiPad/32, NT = Co/16.
//   element e = 2v+pb holds W[kc*32 + (v>>2)*16 + (lane>>4)*8 + (v&3)*2 + pb]
//                            [nt*16 + (lane&15)], zero for K >= CiSrc.
__global__ void k_repack_w(const float* __restrict__ W, unsigned short* __restrict__ Wf,
                           int T, int CiPad, int CiSrc, int Co) {
    long long total = (long long)T * CiPad * Co;
    long long i = (long long)blockIdx.x * blockDim.x + threadIdx.x;
    if (i >= total) return;
    int e = (int)(i & 15);
    long long r = i >> 4;
    int lane = (int)(r & 31); r >>= 5;
    int NT = Co >> 4;
    int nt = (int)(r % NT);  r /= NT;
    int KC = CiPad >> 5;
    int kc = (int)(r % KC);
    int t  = (int)(r / KC);
    int v = e >> 1, pb = e & 1;
    int kk  = kc * 32 + ((v >> 2) * 16) + ((lane >> 4) * 8) + ((v & 3) * 2) + pb;
    int col = nt * 16 + (lane & 15);
    float val = (kk < CiSrc) ? W[((size_t)t * CiSrc + kk) * Co + col] : 0.f;
    Wf[i] = f2bf(val);
}

// transpose-deconv -> 8-tap gather map: nbr[k][i] = (off[i]==k) ? par[i] : sentinel
__global__ void k_make_deconv_nbr(const int* __restrict__ par, const int* __restrict__ off,
                                  int* __restrict__ nbr, int N, int sent) {
    int i = blockIdx.x * blockDim.x + threadIdx.x;
    if (i >= N) return;
    int p = par[i], o = off[i];
#pragma unroll
    for (int k = 0; k < 8; ++k)
        nbr[(size_t)k * N + i] = (o == k) ? p : sent;
}

// ------------------------------ batch norm ---------------------------------

// stats: [sum | sumsq]  ->  [scale | shift]
__global__ void k_bn_finalize(float* __restrict__ stats, const float* __restrict__ g,
                              const float* __restrict__ b, int N, int C) {
    int c = threadIdx.x;
    if (c >= C) return;
    float inv = 1.f / (float)N;
    float mu  = stats[c] * inv;
    float var = stats[C + c] * inv - mu * mu;
    float sc  = g[c] * rsqrtf(var + 1e-5f);
    stats[c]     = sc;
    stats[C + c] = b[c] - mu * sc;
}

__global__ void k_bn_apply(const float* __restrict__ x, const float* __restrict__ stats,
                           const float* __restrict__ add, float* __restrict__ outf,
                           unsigned short* __restrict__ outb, int N, int C, int relu) {
    long long i = (long long)blockIdx.x * blockDim.x + threadIdx.x;
    long long tot = (long long)N * C;
    if (i >= tot) return;
    int c = (int)(i % C);
    float v = x[i] * stats[c] + stats[C + c];
    if (add)  v += add[i];
    if (relu) v = fmaxf(v, 0.f);
    if (outf) outf[i] = v;
    if (outb) outb[i] = f2bf(v);
}

__global__ void k_copy_cols(const float* __restrict__ src, int N, int Cs,
                            float* __restrict__ dstf, unsigned short* __restrict__ dstb,
                            int Cd, int coff) {
    long long i = (long long)blockIdx.x * blockDim.x + threadIdx.x;
    long long tot = (long long)N * Cs;
    if (i >= tot) return;
    int r = (int)(i / Cs);
    int c = (int)(i - (long long)r * Cs);
    float v = src[i];
    size_t o = (size_t)r * Cd + coff + c;
    if (dstf) dstf[o] = v;
    if (dstb) dstb[o] = f2bf(v);
}

// ------------------------------ WMMA sparse conv ---------------------------
//
//   Y[Nout,Co] = sum_t  Xb[ nbr[t][:] ] @ W[t]      (sentinel / OOB row -> 0)
//   nbr == nullptr  => identity gather (dense 1x1 conv), T = 1.
//
// Block = 64 rows; each of the 4 waves owns 16 rows x (NW*16) cols, so one
// gathered A fragment feeds NW WMMAs (amortizes the random-access gather).
// The tap loop is software-pipelined: while tap t's WMMAs run, tap t+1's
// neighbor index is loaded and its row prefetched (global_prefetch).
// Epilogue fuses BN column sum/sumsq via 2 global atomics per lane per tile
// (every accumulator value in a lane belongs to a single output column).
template <int CI, int NW>
__global__ __launch_bounds__(128) void k_spconv_wmma(
    const unsigned short* __restrict__ Xb,   // Nin x CI (bf16 bits)
    const int* __restrict__ nbr,             // T x Nout, or nullptr
    const unsigned short* __restrict__ Wf,   // prepacked [T][CI/32][Co/16][32][16]
    float* __restrict__ Y,                   // Nout x Co
    float* __restrict__ stats,               // 2*Co: sums | sumsq (pre-zeroed)
    int Nout, int Nin, int Co, int T)
{
    constexpr int KC = CI / 32;
    const int lane  = threadIdx.x & 31;
    const int wave  = threadIdx.x >> 5;
    const int row0  = blockIdx.x * 64 + wave * 16;
    const int nt0   = blockIdx.y * NW;
    const int m     = lane & 15;
    const int khalf = lane >> 4;
    const int gRow  = row0 + m;
    const int NT    = Co >> 4;

    // tap-0 gather index
    int src = Nin;
    if (gRow < Nout) src = nbr ? nbr[gRow] : gRow;

    v8f acc[NW] = {};
    for (int t = 0; t < T; ++t) {
        // pipeline: fetch next tap's index + prefetch its row into the caches
        int nsrc = Nin;
        if (t + 1 < T) {
            if (gRow < Nout) nsrc = nbr ? nbr[(size_t)(t + 1) * Nout + gRow] : gRow;
            if ((unsigned)nsrc < (unsigned)Nin)
                __builtin_prefetch(Xb + (size_t)nsrc * CI, 0, 3);
        }

        const bool valid = (unsigned)src < (unsigned)Nin;
        const uint4* xrow  = (const uint4*)(Xb + (size_t)src * CI);
        const uint4* wbase = (const uint4*)Wf +
                             ((((size_t)t * KC) * NT + nt0) * 32 + lane) * 2;
#pragma unroll
        for (int kc = 0; kc < KC; ++kc) {
            uint4 a0{0u, 0u, 0u, 0u}, a1{0u, 0u, 0u, 0u};
            if (valid) {
                a0 = xrow[kc * 4 + khalf];          // K = kc*32 + khalf*8 .. +7
                a1 = xrow[kc * 4 + 2 + khalf];      // K = kc*32 + 16 + khalf*8 ..
            }
            BF16Frag A;
            A.u[0] = a0.x; A.u[1] = a0.y; A.u[2] = a0.z; A.u[3] = a0.w;
            A.u[4] = a1.x; A.u[5] = a1.y; A.u[6] = a1.z; A.u[7] = a1.w;
            const uint4* wkc = wbase + (size_t)kc * NT * 64;
#pragma unroll
            for (int w = 0; w < NW; ++w) {
                uint4 b0 = wkc[w * 64];
                uint4 b1 = wkc[w * 64 + 1];
                BF16Frag B;
                B.u[0] = b0.x; B.u[1] = b0.y; B.u[2] = b0.z; B.u[3] = b0.w;
                B.u[4] = b1.x; B.u[5] = b1.y; B.u[6] = b1.z; B.u[7] = b1.w;
                acc[w] = __builtin_amdgcn_wmma_f32_16x16x32_bf16(
                    false, A.v, false, B.v, (short)0, acc[w], false, false);
            }
        }
        src = nsrc;
    }

    // epilogue: store + fused BN column sums (C layout: col = lane&15 fixed)
#pragma unroll
    for (int w = 0; w < NW; ++w) {
        int nCol = (nt0 + w) * 16 + m;
        float s = 0.f, sq = 0.f;
#pragma unroll
        for (int r = 0; r < 8; ++r) {
            int row = row0 + r + khalf * 8;
            if (row < Nout) {
                float v = acc[w][r];
                Y[(size_t)row * Co + nCol] = v;
                s  += v;
                sq += v * v;
            }
        }
        if (stats) {
            atomicAdd(&stats[nCol], s);
            atomicAdd(&stats[Co + nCol], sq);
        }
    }
}

// ------------------------------ classifier ---------------------------------

__global__ void k_classifier(const float* __restrict__ x, const float* __restrict__ W,
                             const float* __restrict__ b, float* __restrict__ out,
                             int N, int Ci, int Co) {
    long long i = (long long)blockIdx.x * blockDim.x + threadIdx.x;
    if (i >= (long long)N * Co) return;
    int row = (int)(i / Co);
    int c   = (int)(i - (long long)row * Co);
    float s = b[c];
    const float* xr = x + (size_t)row * Ci;
    for (int k = 0; k < Ci; ++k) s += xr[k] * W[(size_t)k * Co + c];
    out[(size_t)row * Co + c] = s;
}

// ===========================================================================

extern "C" void kernel_launch(void* const* d_in, const int* in_sizes, int n_in,
                              void* d_out, int out_size, void* d_ws, size_t ws_size,
                              hipStream_t stream) {
    (void)n_in; (void)out_size; (void)ws_size;

    // -------- workspace bump allocator --------
    char* wsbase = (char*)d_ws;
    size_t wsoff = 0;
    auto alloc = [&](size_t bytes) -> void* {
        wsoff = (wsoff + 255) & ~(size_t)255;
        void* p = wsbase + wsoff;
        wsoff += bytes;
        return p;
    };

    // -------- parse inputs (insertion order; params in jax pytree order) ----
    int ix = 0;
    const float* feats = (const float*)d_in[ix];
    long long featsN = in_sizes[ix]; ix++; (void)featsN;

    auto take_f = [&]() -> const float* { return (const float*)d_in[ix++]; };
    auto take_w = [&](int T, int CiPad, int CiSrc, int Co) -> const unsigned short* {
        const float* w = (const float*)d_in[ix]; ix++;
        long long n = (long long)T * CiPad * Co;
        unsigned short* wf = (unsigned short*)alloc((size_t)n * 2);
        k_repack_w<<<dim3((unsigned)((n + 255) / 256)), dim3(256), 0, stream>>>(
            w, wf, T, CiPad, CiSrc, Co);
        return wf;
    };

    struct ResP  { const unsigned short *W1b, *W2b, *Wdb;
                   const float *b1, *b2, *bd, *g1, *g2, *gd; bool hasWd; };
    struct StgP  { const unsigned short* dWb; const float *db, *dg; ResP r0, r1; };
    struct UpP   { ResP r0, r1; const unsigned short* uWb; const float *ub, *ug; };

    auto take_res = [&](int Ci, int Co, bool hasWd) -> ResP {
        ResP r{}; r.hasWd = hasWd;
        r.W1b = take_w(27, Ci, Ci, Co);
        r.W2b = take_w(27, Co, Co, Co);
        if (hasWd) r.Wdb = take_w(1, Ci, Ci, Co);
        r.b1 = take_f(); r.b2 = take_f();
        if (hasWd) r.bd = take_f();
        r.g1 = take_f(); r.g2 = take_f();
        if (hasWd) r.gd = take_f();
        return r;
    };
    auto take_stage = [&](int cd, int cOut, bool r0HasWd) -> StgP {
        StgP s{};
        s.dWb = take_w(8, cd, cd, cd);
        s.db = take_f(); s.dg = take_f();
        s.r0 = take_res(cd, cOut, r0HasWd);
        s.r1 = take_res(cOut, cOut, false);
        return s;
    };
    auto take_up = [&](int skipC) -> UpP {
        UpP u{};
        u.r0 = take_res(256 + skipC, 256, true);
        u.r1 = take_res(256, 256, false);
        u.uWb = take_w(8, 256, 256, 256);
        u.ub = take_f(); u.ug = take_f();
        return u;
    };

    // params pytree (dict keys sorted; 'W' < 'b' < 'g')
    const float* clsW = take_f();            // 256 x 17, stays fp32
    const float* clsB = take_f();
    StgP st[4];
    st[0] = take_stage(32, 32, false);
    st[1] = take_stage(32, 64, true);
    st[2] = take_stage(64, 128, true);
    st[3] = take_stage(128, 256, true);
    const unsigned short* stemWb = take_w(27, 32, 4, 32);   // pad Ci 4 -> 32
    const float* stemB = take_f();
    const float* stemG = take_f();
    UpP up[4];
    up[0] = take_up(128);
    up[1] = take_up(64);
    up[2] = take_up(32);
    up[3] = take_up(32);

    // maps
    const int* nbr[5]; int Nl[5];
    for (int l = 0; l < 5; ++l) { nbr[l] = (const int*)d_in[ix]; Nl[l] = in_sizes[ix] / 27; ix++; }
    const int* dn[4];
    for (int i = 0; i < 4; ++i) { dn[i] = (const int*)d_in[ix]; ix++; }
    const int* upPar[4]; const int* upOff[4];
    for (int i = 0; i < 4; ++i) { upPar[i] = (const int*)d_in[ix]; ix++;
                                  upOff[i] = (const int*)d_in[ix]; ix++; }

    const int N0 = Nl[0];
    const int Cs[5] = {32, 32, 64, 128, 256};

    // -------- activation / scratch buffers --------
    float*          xsf[5]; unsigned short* xsb[5];
    for (int l = 0; l < 5; ++l) {
        xsf[l] = (float*)alloc((size_t)Nl[l] * Cs[l] * 4);
        xsb[l] = (unsigned short*)alloc((size_t)Nl[l] * Cs[l] * 2);
    }
    unsigned short* featsb = (unsigned short*)alloc((size_t)N0 * 32 * 2);  // padded
    float* tA = (float*)alloc((size_t)N0 * 256 * 4);
    float* tB = (float*)alloc((size_t)N0 * 256 * 4);
    float* tC = (float*)alloc((size_t)N0 * 256 * 4);
    unsigned short* tAb = (unsigned short*)alloc((size_t)N0 * 256 * 2);
    float*          cur0f = (float*)alloc((size_t)N0 * 384 * 4);
    unsigned short* cur0b = (unsigned short*)alloc((size_t)N0 * 384 * 2);
    float*          cur1f = (float*)alloc((size_t)N0 * 384 * 4);
    unsigned short* cur1b = (unsigned short*)alloc((size_t)N0 * 384 * 2);
    int*   nbrtmp = (int*)alloc((size_t)8 * N0 * 4);
    float* stats  = (float*)alloc(2 * 384 * 4);

    // -------- host-side building blocks --------
    auto spconv = [&](const unsigned short* xb, const int* nb, const unsigned short* wf,
                      float* y, int Nout, int Nin, int Ci, int Co, int T) {
        const int NW = (Co == 32) ? 2 : 4;
        dim3 grid((Nout + 63) / 64, (Co >> 4) / NW);
        dim3 blk(128);
#define LAUNCH_CI(CI_)                                                                     \
        do { if (NW == 2)                                                                  \
                 k_spconv_wmma<CI_, 2><<<grid, blk, 0, stream>>>(xb, nb, wf, y, stats,     \
                                                                 Nout, Nin, Co, T);        \
             else                                                                          \
                 k_spconv_wmma<CI_, 4><<<grid, blk, 0, stream>>>(xb, nb, wf, y, stats,     \
                                                                 Nout, Nin, Co, T);        \
        } while (0)
        switch (Ci) {
        case 32:  LAUNCH_CI(32);  break;
        case 64:  LAUNCH_CI(64);  break;
        case 128: LAUNCH_CI(128); break;
        case 256: LAUNCH_CI(256); break;
        case 288: LAUNCH_CI(288); break;
        case 320: LAUNCH_CI(320); break;
        case 384: LAUNCH_CI(384); break;
        default:  break;
        }
#undef LAUNCH_CI
    };
    // conv + fused BN stats + finalize (stats buffer zeroed first)
    auto convbn = [&](const unsigned short* xb, const int* nb, const unsigned short* wf,
                      float* y, int Nout, int Nin, int Ci, int Co, int T,
                      const float* g, const float* b) {
        k_zero<<<dim3((2 * Co + 255) / 256), dim3(256), 0, stream>>>(stats, 2 * Co);
        spconv(xb, nb, wf, y, Nout, Nin, Ci, Co, T);
        k_bn_finalize<<<dim3(1), dim3(Co), 0, stream>>>(stats, g, b, Nout, Co);
    };
    auto bnapply = [&](const float* x, const float* add, float* of, unsigned short* ob,
                       int N, int C, bool relu) {
        long long tot = (long long)N * C;
        k_bn_apply<<<dim3((unsigned)((tot + 255) / 256)), dim3(256), 0, stream>>>(
            x, stats, add, of, ob, N, C, relu ? 1 : 0);
    };
    auto copycols = [&](const float* src, int N, int Csrc, float* df, unsigned short* db,
                        int Cd, int coff) {
        long long tot = (long long)N * Csrc;
        k_copy_cols<<<dim3((unsigned)((tot + 255) / 256)), dim3(256), 0, stream>>>(
            src, N, Csrc, df, db, Cd, coff);
    };
    auto resblock = [&](const float* xf, const unsigned short* xb, int N, const int* nb,
                        const ResP& R, int Ci, int Co, float* outf, unsigned short* outb) {
        // h1 = relu(bn(conv27(x, W1)))
        convbn(xb, nb, R.W1b, tA, N, N, Ci, Co, 27, R.g1, R.b1);
        bnapply(tA, nullptr, nullptr, tAb, N, Co, true);
        // shortcut sc = bn(x @ Wd) or x
        const float* sc = xf;
        if (R.hasWd) {
            convbn(xb, nullptr, R.Wdb, tC, N, N, Ci, Co, 1, R.gd, R.bd);
            bnapply(tC, nullptr, tC, nullptr, N, Co, false);   // in-place elementwise
            sc = tC;
        }
        // out = relu(bn(conv27(h1, W2)) + sc)
        convbn(tAb, nb, R.W2b, tB, N, N, Co, Co, 27, R.g2, R.b2);
        bnapply(tB, sc, outf, outb, N, Co, true);
    };

    // -------- forward pass --------
    // stem: x0 = relu(bn(conv27(feats)))   (feats padded to 32 channels)
    {
        long long n = (long long)N0 * 32;
        k_cvt_pad<<<dim3((unsigned)((n + 255) / 256)), dim3(256), 0, stream>>>(
            feats, featsb, n, 4);
    }
    convbn(featsb, nbr[0], stemWb, tA, N0, N0, 32, 32, 27, stemG, stemB);
    bnapply(tA, nullptr, xsf[0], xsb[0], N0, 32, true);

    // encoder
    for (int i = 1; i <= 4; ++i) {
        const StgP& S = st[i - 1];
        int Cprev = Cs[i - 1];
        convbn(xsb[i - 1], dn[i - 1], S.dWb, tA, Nl[i], Nl[i - 1], Cprev, Cprev, 8,
               S.dg, S.db);
        bnapply(tA, nullptr, cur0f, cur0b, Nl[i], Cprev, true);
        resblock(cur0f, cur0b, Nl[i], nbr[i], S.r0, Cprev, Cs[i], cur1f, cur1b);
        resblock(cur1f, cur1b, Nl[i], nbr[i], S.r1, Cs[i], Cs[i], xsf[i], xsb[i]);
    }

    // decoder
    const float* yf = xsf[4];
    const unsigned short* yb = xsb[4];
    int yN = Nl[4];
    for (int u = 0; u < 4; ++u) {
        int lf = 3 - u;
        int Nf = Nl[lf];
        int Nc = yN;
        k_make_deconv_nbr<<<dim3((Nf + 255) / 256), dim3(256), 0, stream>>>(
            upPar[u], upOff[u], nbrtmp, Nf, Nc);
        convbn(yb, nbrtmp, up[u].uWb, tA, Nf, Nc, 256, 256, 8, up[u].ug, up[u].ub);
        bnapply(tA, nullptr, tB, nullptr, Nf, 256, true);
        // concat [deconv | skip]
        int skipC = Cs[lf];
        int Cc = 256 + skipC;
        copycols(tB, Nf, 256, cur0f, cur0b, Cc, 0);
        copycols(xsf[lf], Nf, skipC, cur0f, cur0b, Cc, 256);
        // res blocks
        resblock(cur0f, cur0b, Nf, nbr[lf], up[u].r0, Cc, 256, cur1f, cur1b);
        resblock(cur1f, cur1b, Nf, nbr[lf], up[u].r1, 256, 256, cur0f, cur0b);
        yf = cur0f; yb = cur0b; yN = Nf;
    }

    // classifier: out = y @ W + b   (N0 x 17, fp32)
    long long tot = (long long)N0 * 17;
    k_classifier<<<dim3((unsigned)((tot + 255) / 256)), dim3(256), 0, stream>>>(
        yf, clsW, clsB, (float*)d_out, N0, 256, 17);
}
